// LlamaDecoderLayer_89867895701680
// MI455X (gfx1250) — compile-verified, matrix operand order checked by reading, not run on hardware
//
#include <hip/hip_runtime.h>
#include <math.h>
#include <stdint.h>

// ---------------- problem constants (match reference) ----------------
#define HIDDEN_DIM 2048
#define INTER_DIM  8192
#define NHEADS     16
#define HEAD_DIM   128
#define SEQ        2048
#define NBATCH     2
#define MTOK       (NBATCH * SEQ)   // 4096 tokens

// ---------------- CDNA5 WMMA types ----------------
typedef __attribute__((ext_vector_type(16))) __bf16 v16bf;
typedef __attribute__((ext_vector_type(8)))  float  v8f;

union FragB { v16bf v; uint4 q[2]; };
union AccF  { v8f v; float f[8]; };

__device__ __forceinline__ unsigned short f2bf(float x) {
  union { float f; uint32_t u; } cv; cv.f = x;
  uint32_t u = cv.u;
  u += 0x7FFFu + ((u >> 16) & 1u);   // round to nearest even
  return (unsigned short)(u >> 16);
}

// pack two f32 -> packed bf16 pair (lo in [15:0], hi in [31:16])
__device__ __forceinline__ uint32_t pack2bf(float lo, float hi) {
#if __has_builtin(__builtin_amdgcn_cvt_pk_bf16_f32)
  auto r = __builtin_amdgcn_cvt_pk_bf16_f32(lo, hi);
  uint32_t u; __builtin_memcpy(&u, &r, sizeof(u));
  return u;
#elif __has_builtin(__builtin_amdgcn_perm)
  // round-to-nearest (ties away), then one v_perm_b32 packs both high halves
  union { float f; uint32_t u; } a, b;
  a.f = lo; b.f = hi;
  return __builtin_amdgcn_perm(b.u + 0x8000u, a.u + 0x8000u, 0x07060302u);
#else
  return (uint32_t)f2bf(lo) | ((uint32_t)f2bf(hi) << 16);
#endif
}

// CDNA5 async copy: 16B global -> LDS per lane, tracked on ASYNCcnt.
// VDST operand is the wave-relative LDS byte address (= low 32 bits of a
// generic pointer into LDS, per the aperture mapping rules).
__device__ __forceinline__ void async_copy_b128(const unsigned short* __restrict__ g,
                                                unsigned short* l) {
  const uint32_t loff = (uint32_t)(uintptr_t)l;
  asm volatile("global_load_async_to_lds_b128 %0, %1, off"
               :: "v"(loff), "v"((unsigned long long)(uintptr_t)g)
               : "memory");
}
__device__ __forceinline__ void wait_async0() {
#if __has_builtin(__builtin_amdgcn_s_wait_asynccnt)
  __builtin_amdgcn_s_wait_asynccnt(0);
#else
  asm volatile("s_wait_asynccnt 0x0" ::: "memory");
#endif
}

// Load a 16x32 bf16 fragment (A layout; B^T layout identical with n as "row").
// CDNA5 16-bit A layout: lane(0..15)=row, lane/16 selects K halves:
//   h[0..7]  <- K = k_base + g*8 + 0..7
//   h[8..15] <- K = k_base + 16 + g*8 + 0..7
__device__ __forceinline__ v16bf load_frag_lds(const unsigned short* base, int stride,
                                               int row_base, int k_base) {
  const int lane = threadIdx.x & 31;
  const int g = lane >> 4;
  const int r = lane & 15;
  const unsigned short* p = base + (size_t)(row_base + r) * stride + k_base + g * 8;
  FragB f;
  f.q[0] = *(const uint4*)(p);
  f.q[1] = *(const uint4*)(p + 16);
  return f.v;
}

__device__ __forceinline__ v8f wmma_bf16(v16bf a, v16bf b, v8f c) {
  return __builtin_amdgcn_wmma_f32_16x16x32_bf16(false, a, false, b, (short)0, c,
                                                 false, false);
}

// ---------------- RMSNorm (fp32 in) -> bf16 normalized*weight ----------------
__global__ void __launch_bounds__(256)
rmsnorm_to_bf16_kernel(const float* __restrict__ x, const float* __restrict__ w,
                       unsigned short* __restrict__ out) {
  const int row = blockIdx.x;
  const int tid = threadIdx.x;
  const float* xr = x + (size_t)row * HIDDEN_DIM;
  float ss = 0.f;
#pragma unroll
  for (int base = 0; base < HIDDEN_DIM; base += 256 * 4) {
    float4 v = *(const float4*)(xr + base + tid * 4);
    ss += v.x * v.x + v.y * v.y + v.z * v.z + v.w * v.w;
  }
#pragma unroll
  for (int off = 16; off > 0; off >>= 1) ss += __shfl_xor(ss, off);
  __shared__ float red[8];
  if ((tid & 31) == 0) red[tid >> 5] = ss;
  __syncthreads();
  float tot = 0.f;
#pragma unroll
  for (int i = 0; i < 8; ++i) tot += red[i];
  const float inv = rsqrtf(tot / (float)HIDDEN_DIM + 1e-6f);
#pragma unroll
  for (int base = 0; base < HIDDEN_DIM; base += 256 * 4) {
    const int c = base + tid * 4;
    float4 v  = *(const float4*)(xr + c);
    float4 wv = *(const float4*)(w + c);
    union { uint32_t u[2]; uint2 u2; } pk;
    pk.u[0] = pack2bf(v.x * inv * wv.x, v.y * inv * wv.y);
    pk.u[1] = pack2bf(v.z * inv * wv.z, v.w * inv * wv.w);
    *(uint2*)(out + (size_t)row * HIDDEN_DIM + c) = pk.u2;
  }
}

// ---------------- GEMM: C[M,N] = A(bf16)[M,K] * B(f32 weights)[K,N] ----------
// Tile 128x128x32, 256 threads = 8 waves (4 in M, 2 in N), wave tile 32x64.
// Double-buffered LDS: A staged with async-to-LDS, B prefetched in registers
// (f32->bf16 packed convert + transpose, b32 LDS stores). One barrier/K-step.
// EPI 0: store f32   EPI 1: store bf16   EPI 2: f32 out = aux + C (residual)
template<int EPI>
__global__ void __launch_bounds__(256)
gemm_bf16_kernel(const unsigned short* __restrict__ A, const float* __restrict__ Bw,
                 void* __restrict__ outP, const float* __restrict__ aux,
                 int M, int N, int K) {
  __shared__ unsigned short sA [2][128 * 40];   // A[m][k], row stride 40 halves
  __shared__ unsigned short sBT[2][128 * 40];   // B^T[n][k], row stride 40 halves
  const int tid  = threadIdx.x;
  const int wave = tid >> 5;
  const int wm = wave & 3;   // 0..3 (M)
  const int wn = wave >> 2;  // 0..1 (N)
  const int m0 = blockIdx.y * 128;
  const int n0 = blockIdx.x * 128;

  // per-thread staging coordinates
  const int ar0 = tid >> 2;                 // A row (2 passes of +64)
  const int acg = (tid & 3) * 8;            // A col group (8 halves)

  AccF acc[2][4];
#pragma unroll
  for (int a = 0; a < 2; ++a)
#pragma unroll
    for (int b = 0; b < 4; ++b)
#pragma unroll
      for (int i = 0; i < 8; ++i) acc[a][b].f[i] = 0.f;

  // ---- prologue: stage tile 0 into buffer 0 ----
#pragma unroll
  for (int pass = 0; pass < 2; ++pass) {
    const int r = ar0 + pass * 64;
    async_copy_b128(A + (size_t)(m0 + r) * K + acg, &sA[0][r * 40 + acg]);
  }
#pragma unroll
  for (int pass = 0; pass < 2; ++pass) {
    const int idx = pass * 256 + tid;
    const int kr = (idx >> 5) * 2;        // even k row
    const int ng = (idx & 31) * 4;
    float4 r0 = *(const float4*)(Bw + (size_t)kr * N + n0 + ng);
    float4 r1 = *(const float4*)(Bw + (size_t)(kr + 1) * N + n0 + ng);
    *(uint32_t*)&sBT[0][(ng + 0) * 40 + kr] = pack2bf(r0.x, r1.x);
    *(uint32_t*)&sBT[0][(ng + 1) * 40 + kr] = pack2bf(r0.y, r1.y);
    *(uint32_t*)&sBT[0][(ng + 2) * 40 + kr] = pack2bf(r0.z, r1.z);
    *(uint32_t*)&sBT[0][(ng + 3) * 40 + kr] = pack2bf(r0.w, r1.w);
  }
  wait_async0();
  __syncthreads();

  const int nk = K >> 5;
  for (int kt = 0; kt < nk; ++kt) {
    const int cur = kt & 1, nxt = cur ^ 1;
    const bool more = (kt + 1) < nk;
    float4 p0[2], p1[2];
    if (more) {
      const int k1 = (kt + 1) * 32;
#pragma unroll
      for (int pass = 0; pass < 2; ++pass) {
        const int r = ar0 + pass * 64;
        async_copy_b128(A + (size_t)(m0 + r) * K + k1 + acg,
                        &sA[nxt][r * 40 + acg]);
      }
#pragma unroll
      for (int pass = 0; pass < 2; ++pass) {
        const int idx = pass * 256 + tid;
        const int kr = (idx >> 5) * 2;
        const int ng = (idx & 31) * 4;
        p0[pass] = *(const float4*)(Bw + (size_t)(k1 + kr) * N + n0 + ng);
        p1[pass] = *(const float4*)(Bw + (size_t)(k1 + kr + 1) * N + n0 + ng);
      }
    }

    // ---- compute on current buffer: batch frag loads, then 8 wmmas ----
    const unsigned short* cA = sA[cur];
    const unsigned short* cB = sBT[cur];
    v16bf af0 = load_frag_lds(cA, 40, wm * 32 + 0,  0);
    v16bf af1 = load_frag_lds(cA, 40, wm * 32 + 16, 0);
    v16bf bf0 = load_frag_lds(cB, 40, wn * 64 + 0,  0);
    v16bf bf1 = load_frag_lds(cB, 40, wn * 64 + 16, 0);
    v16bf bf2 = load_frag_lds(cB, 40, wn * 64 + 32, 0);
    v16bf bf3 = load_frag_lds(cB, 40, wn * 64 + 48, 0);
    acc[0][0].v = wmma_bf16(af0, bf0, acc[0][0].v);
    acc[1][0].v = wmma_bf16(af1, bf0, acc[1][0].v);
    acc[0][1].v = wmma_bf16(af0, bf1, acc[0][1].v);
    acc[1][1].v = wmma_bf16(af1, bf1, acc[1][1].v);
    acc[0][2].v = wmma_bf16(af0, bf2, acc[0][2].v);
    acc[1][2].v = wmma_bf16(af1, bf2, acc[1][2].v);
    acc[0][3].v = wmma_bf16(af0, bf3, acc[0][3].v);
    acc[1][3].v = wmma_bf16(af1, bf3, acc[1][3].v);

    if (more) {
#pragma unroll
      for (int pass = 0; pass < 2; ++pass) {
        const int idx = pass * 256 + tid;
        const int kr = (idx >> 5) * 2;
        const int ng = (idx & 31) * 4;
        *(uint32_t*)&sBT[nxt][(ng + 0) * 40 + kr] = pack2bf(p0[pass].x, p1[pass].x);
        *(uint32_t*)&sBT[nxt][(ng + 1) * 40 + kr] = pack2bf(p0[pass].y, p1[pass].y);
        *(uint32_t*)&sBT[nxt][(ng + 2) * 40 + kr] = pack2bf(p0[pass].z, p1[pass].z);
        *(uint32_t*)&sBT[nxt][(ng + 3) * 40 + kr] = pack2bf(p0[pass].w, p1[pass].w);
      }
      wait_async0();
    }
    __syncthreads();
  }

  const int lane = tid & 31, g = lane >> 4, c15 = lane & 15;
#pragma unroll
  for (int ms = 0; ms < 2; ++ms)
#pragma unroll
    for (int nt = 0; nt < 4; ++nt)
#pragma unroll
      for (int i = 0; i < 8; ++i) {
        const size_t row = (size_t)(m0 + wm * 32 + ms * 16 + g * 8 + i);
        const size_t col = (size_t)(n0 + wn * 64 + nt * 16 + c15);
        const float v = acc[ms][nt].f[i];
        if (EPI == 0) {
          ((float*)outP)[row * N + col] = v;
        } else if (EPI == 1) {
          ((unsigned short*)outP)[row * N + col] = f2bf(v);
        } else {
          ((float*)outP)[row * N + col] = v + aux[row * N + col];
        }
      }
}

// ---------------- fused gate+up GEMM: act = silu(X*Wg) * (X*Wu), bf16 out ----
__global__ void __launch_bounds__(256)
gemm_gateup_kernel(const unsigned short* __restrict__ A,
                   const float* __restrict__ Wg, const float* __restrict__ Wu,
                   unsigned short* __restrict__ outAct, int M, int N, int K) {
  __shared__ unsigned short sA [2][128 * 40];
  __shared__ unsigned short sGT[2][128 * 40];
  __shared__ unsigned short sUT[2][128 * 40];
  const int tid  = threadIdx.x;
  const int wave = tid >> 5;
  const int wm = wave & 3;
  const int wn = wave >> 2;
  const int m0 = blockIdx.y * 128;
  const int n0 = blockIdx.x * 128;

  const int ar0 = tid >> 2;
  const int acg = (tid & 3) * 8;

  AccF accG[2][4], accU[2][4];
#pragma unroll
  for (int a = 0; a < 2; ++a)
#pragma unroll
    for (int b = 0; b < 4; ++b)
#pragma unroll
      for (int i = 0; i < 8; ++i) { accG[a][b].f[i] = 0.f; accU[a][b].f[i] = 0.f; }

  // prologue
#pragma unroll
  for (int pass = 0; pass < 2; ++pass) {
    const int r = ar0 + pass * 64;
    async_copy_b128(A + (size_t)(m0 + r) * K + acg, &sA[0][r * 40 + acg]);
  }
#pragma unroll
  for (int pass = 0; pass < 2; ++pass) {
    const int idx = pass * 256 + tid;
    const int kr = (idx >> 5) * 2;
    const int ng = (idx & 31) * 4;
    float4 g0 = *(const float4*)(Wg + (size_t)kr * N + n0 + ng);
    float4 g1 = *(const float4*)(Wg + (size_t)(kr + 1) * N + n0 + ng);
    float4 u0 = *(const float4*)(Wu + (size_t)kr * N + n0 + ng);
    float4 u1 = *(const float4*)(Wu + (size_t)(kr + 1) * N + n0 + ng);
    *(uint32_t*)&sGT[0][(ng + 0) * 40 + kr] = pack2bf(g0.x, g1.x);
    *(uint32_t*)&sGT[0][(ng + 1) * 40 + kr] = pack2bf(g0.y, g1.y);
    *(uint32_t*)&sGT[0][(ng + 2) * 40 + kr] = pack2bf(g0.z, g1.z);
    *(uint32_t*)&sGT[0][(ng + 3) * 40 + kr] = pack2bf(g0.w, g1.w);
    *(uint32_t*)&sUT[0][(ng + 0) * 40 + kr] = pack2bf(u0.x, u1.x);
    *(uint32_t*)&sUT[0][(ng + 1) * 40 + kr] = pack2bf(u0.y, u1.y);
    *(uint32_t*)&sUT[0][(ng + 2) * 40 + kr] = pack2bf(u0.z, u1.z);
    *(uint32_t*)&sUT[0][(ng + 3) * 40 + kr] = pack2bf(u0.w, u1.w);
  }
  wait_async0();
  __syncthreads();

  const int nk = K >> 5;
  for (int kt = 0; kt < nk; ++kt) {
    const int cur = kt & 1, nxt = cur ^ 1;
    const bool more = (kt + 1) < nk;
    float4 pg0[2], pg1[2], pu0[2], pu1[2];
    if (more) {
      const int k1 = (kt + 1) * 32;
#pragma unroll
      for (int pass = 0; pass < 2; ++pass) {
        const int r = ar0 + pass * 64;
        async_copy_b128(A + (size_t)(m0 + r) * K + k1 + acg,
                        &sA[nxt][r * 40 + acg]);
      }
#pragma unroll
      for (int pass = 0; pass < 2; ++pass) {
        const int idx = pass * 256 + tid;
        const int kr = (idx >> 5) * 2;
        const int ng = (idx & 31) * 4;
        pg0[pass] = *(const float4*)(Wg + (size_t)(k1 + kr) * N + n0 + ng);
        pg1[pass] = *(const float4*)(Wg + (size_t)(k1 + kr + 1) * N + n0 + ng);
        pu0[pass] = *(const float4*)(Wu + (size_t)(k1 + kr) * N + n0 + ng);
        pu1[pass] = *(const float4*)(Wu + (size_t)(k1 + kr + 1) * N + n0 + ng);
      }
    }

    const unsigned short* cA = sA[cur];
    v16bf af0 = load_frag_lds(cA, 40, wm * 32 + 0,  0);
    v16bf af1 = load_frag_lds(cA, 40, wm * 32 + 16, 0);
    {
      const unsigned short* cG = sGT[cur];
      v16bf b0 = load_frag_lds(cG, 40, wn * 64 + 0,  0);
      v16bf b1 = load_frag_lds(cG, 40, wn * 64 + 16, 0);
      v16bf b2 = load_frag_lds(cG, 40, wn * 64 + 32, 0);
      v16bf b3 = load_frag_lds(cG, 40, wn * 64 + 48, 0);
      accG[0][0].v = wmma_bf16(af0, b0, accG[0][0].v);
      accG[1][0].v = wmma_bf16(af1, b0, accG[1][0].v);
      accG[0][1].v = wmma_bf16(af0, b1, accG[0][1].v);
      accG[1][1].v = wmma_bf16(af1, b1, accG[1][1].v);
      accG[0][2].v = wmma_bf16(af0, b2, accG[0][2].v);
      accG[1][2].v = wmma_bf16(af1, b2, accG[1][2].v);
      accG[0][3].v = wmma_bf16(af0, b3, accG[0][3].v);
      accG[1][3].v = wmma_bf16(af1, b3, accG[1][3].v);
    }
    {
      const unsigned short* cU = sUT[cur];
      v16bf b0 = load_frag_lds(cU, 40, wn * 64 + 0,  0);
      v16bf b1 = load_frag_lds(cU, 40, wn * 64 + 16, 0);
      v16bf b2 = load_frag_lds(cU, 40, wn * 64 + 32, 0);
      v16bf b3 = load_frag_lds(cU, 40, wn * 64 + 48, 0);
      accU[0][0].v = wmma_bf16(af0, b0, accU[0][0].v);
      accU[1][0].v = wmma_bf16(af1, b0, accU[1][0].v);
      accU[0][1].v = wmma_bf16(af0, b1, accU[0][1].v);
      accU[1][1].v = wmma_bf16(af1, b1, accU[1][1].v);
      accU[0][2].v = wmma_bf16(af0, b2, accU[0][2].v);
      accU[1][2].v = wmma_bf16(af1, b2, accU[1][2].v);
      accU[0][3].v = wmma_bf16(af0, b3, accU[0][3].v);
      accU[1][3].v = wmma_bf16(af1, b3, accU[1][3].v);
    }

    if (more) {
#pragma unroll
      for (int pass = 0; pass < 2; ++pass) {
        const int idx = pass * 256 + tid;
        const int kr = (idx >> 5) * 2;
        const int ng = (idx & 31) * 4;
        *(uint32_t*)&sGT[nxt][(ng + 0) * 40 + kr] = pack2bf(pg0[pass].x, pg1[pass].x);
        *(uint32_t*)&sGT[nxt][(ng + 1) * 40 + kr] = pack2bf(pg0[pass].y, pg1[pass].y);
        *(uint32_t*)&sGT[nxt][(ng + 2) * 40 + kr] = pack2bf(pg0[pass].z, pg1[pass].z);
        *(uint32_t*)&sGT[nxt][(ng + 3) * 40 + kr] = pack2bf(pg0[pass].w, pg1[pass].w);
        *(uint32_t*)&sUT[nxt][(ng + 0) * 40 + kr] = pack2bf(pu0[pass].x, pu1[pass].x);
        *(uint32_t*)&sUT[nxt][(ng + 1) * 40 + kr] = pack2bf(pu0[pass].y, pu1[pass].y);
        *(uint32_t*)&sUT[nxt][(ng + 2) * 40 + kr] = pack2bf(pu0[pass].z, pu1[pass].z);
        *(uint32_t*)&sUT[nxt][(ng + 3) * 40 + kr] = pack2bf(pu0[pass].w, pu1[pass].w);
      }
      wait_async0();
    }
    __syncthreads();
  }

  const int lane = tid & 31, g = lane >> 4, c15 = lane & 15;
#pragma unroll
  for (int ms = 0; ms < 2; ++ms)
#pragma unroll
    for (int nt = 0; nt < 4; ++nt)
#pragma unroll
      for (int i = 0; i < 8; ++i) {
        const size_t row = (size_t)(m0 + wm * 32 + ms * 16 + g * 8 + i);
        const size_t col = (size_t)(n0 + wn * 64 + nt * 16 + c15);
        const float gv = accG[ms][nt].f[i];
        const float uv = accU[ms][nt].f[i];
        const float act = (gv / (1.f + __expf(-gv))) * uv;   // silu(g)*u
        outAct[row * N + col] = f2bf(act);
      }
}

// ---------------- flash attention: causal softmax(QK^T/sqrt(d)) V -----------
// grid: (SEQ/128, NBATCH*NHEADS), 256 threads = 8 waves, wave owns 16 q-rows.
__global__ void __launch_bounds__(256)
attn_kernel(const unsigned short* __restrict__ Q, const unsigned short* __restrict__ Kb,
            const unsigned short* __restrict__ Vb, unsigned short* __restrict__ O) {
  __shared__ unsigned short sK [64  * 136];  // K[tok][d]  (== B^T for QK^T)
  __shared__ unsigned short sVT[128 * 72];   // V^T[d][tok] (== B^T for P*V)
  __shared__ unsigned short sP [128 * 72];   // P[qrow][tok] bf16
  const int tid  = threadIdx.x;
  const int wave = tid >> 5;
  const int lane = tid & 31, g = lane >> 4, c15 = lane & 15;
  const int bh = blockIdx.y;
  const int b = bh >> 4, h = bh & 15;
  const int qb0 = blockIdx.x * 128;

  // Q fragments for this wave's 16 rows (4 K-steps over head_dim=128)
  v16bf qf[4];
  {
    const size_t qbase =
        ((size_t)(b * SEQ + qb0 + wave * 16 + c15)) * HIDDEN_DIM + h * HEAD_DIM;
#pragma unroll
    for (int ks = 0; ks < 4; ++ks) {
      FragB f;
      f.q[0] = *(const uint4*)(Q + qbase + ks * 32 + g * 8);
      f.q[1] = *(const uint4*)(Q + qbase + ks * 32 + 16 + g * 8);
      qf[ks] = f.v;
    }
  }

  float mrow[8], lrow[8];
  AccF o[8];
#pragma unroll
  for (int i = 0; i < 8; ++i) { mrow[i] = -1e30f; lrow[i] = 0.f; }
#pragma unroll
  for (int d = 0; d < 8; ++d)
#pragma unroll
    for (int i = 0; i < 8; ++i) o[d].f[i] = 0.f;

  const float scale = 0.08838834764831845f;  // 1/sqrt(128)

  for (int kb0 = 0; kb0 < qb0 + 128; kb0 += 64) {
    __syncthreads();
    // stage K via async-to-LDS (direct copy)
#pragma unroll
    for (int pass = 0; pass < 4; ++pass) {
      const int idx = pass * 256 + tid;
      const int kt = idx >> 4;
      const int dg = (idx & 15) * 8;
      const size_t src =
          ((size_t)(b * SEQ + kb0 + kt)) * HIDDEN_DIM + h * HEAD_DIM + dg;
      async_copy_b128(Kb + src, &sK[kt * 136 + dg]);
    }
    // stage V transposed: token pairs packed into b32 LDS stores
#pragma unroll
    for (int pass = 0; pass < 2; ++pass) {
      const int idx = pass * 256 + tid;
      const int kt = (idx >> 4) * 2;       // even token
      const int dg = (idx & 15) * 8;
      const size_t src =
          ((size_t)(b * SEQ + kb0 + kt)) * HIDDEN_DIM + h * HEAD_DIM + dg;
      uint4 v0 = *(const uint4*)(Vb + src);
      uint4 v1 = *(const uint4*)(Vb + src + HIDDEN_DIM);
      union { uint4 q; unsigned short us[8]; } a, c;
      a.q = v0; c.q = v1;
#pragma unroll
      for (int j = 0; j < 8; ++j) {
        const uint32_t pk = (uint32_t)a.us[j] | ((uint32_t)c.us[j] << 16);
        *(uint32_t*)&sVT[(dg + j) * 72 + kt] = pk;
      }
    }
    wait_async0();
    __syncthreads();

    // S = Q K^T for 16 q-rows x 64 keys
    AccF s[4];
#pragma unroll
    for (int nt = 0; nt < 4; ++nt)
#pragma unroll
      for (int i = 0; i < 8; ++i) s[nt].f[i] = 0.f;
#pragma unroll
    for (int ks = 0; ks < 4; ++ks) {
      v16bf k0 = load_frag_lds(sK, 136, 0,  ks * 32);
      v16bf k1 = load_frag_lds(sK, 136, 16, ks * 32);
      v16bf k2 = load_frag_lds(sK, 136, 32, ks * 32);
      v16bf k3 = load_frag_lds(sK, 136, 48, ks * 32);
      s[0].v = wmma_bf16(qf[ks], k0, s[0].v);
      s[1].v = wmma_bf16(qf[ks], k1, s[1].v);
      s[2].v = wmma_bf16(qf[ks], k2, s[2].v);
      s[3].v = wmma_bf16(qf[ks], k3, s[3].v);
    }

    // causal mask + online softmax; write P tile (bf16) to LDS
#pragma unroll
    for (int i = 0; i < 8; ++i) {
      const int qrow = qb0 + wave * 16 + g * 8 + i;
      float mx = -1e30f;
#pragma unroll
      for (int nt = 0; nt < 4; ++nt) {
        const int kcol = kb0 + nt * 16 + c15;
        float v = s[nt].f[i] * scale;
        if (kcol > qrow) v = -1e30f;
        s[nt].f[i] = v;
        mx = fmaxf(mx, v);
      }
#pragma unroll
      for (int off = 1; off < 16; off <<= 1) mx = fmaxf(mx, __shfl_xor(mx, off));
      const float mnew = fmaxf(mrow[i], mx);
      const float fac = __expf(mrow[i] - mnew);
      mrow[i] = mnew;
      float rs = 0.f;
#pragma unroll
      for (int nt = 0; nt < 4; ++nt) {
        const float p = __expf(s[nt].f[i] - mnew);
        s[nt].f[i] = p;
        rs += p;
      }
#pragma unroll
      for (int off = 1; off < 16; off <<= 1) rs += __shfl_xor(rs, off);
      lrow[i] = lrow[i] * fac + rs;
#pragma unroll
      for (int d = 0; d < 8; ++d) o[d].f[i] *= fac;
      const int prow = wave * 16 + g * 8 + i;
#pragma unroll
      for (int nt = 0; nt < 4; ++nt)
        sP[prow * 72 + nt * 16 + c15] = f2bf(s[nt].f[i]);
    }
    __syncthreads();

    // O += P V
#pragma unroll
    for (int ks2 = 0; ks2 < 2; ++ks2) {
      v16bf af = load_frag_lds(sP, 72, wave * 16, ks2 * 32);
#pragma unroll
      for (int half = 0; half < 2; ++half) {
        v16bf v0 = load_frag_lds(sVT, 72, (half * 4 + 0) * 16, ks2 * 32);
        v16bf v1 = load_frag_lds(sVT, 72, (half * 4 + 1) * 16, ks2 * 32);
        v16bf v2 = load_frag_lds(sVT, 72, (half * 4 + 2) * 16, ks2 * 32);
        v16bf v3 = load_frag_lds(sVT, 72, (half * 4 + 3) * 16, ks2 * 32);
        o[half * 4 + 0].v = wmma_bf16(af, v0, o[half * 4 + 0].v);
        o[half * 4 + 1].v = wmma_bf16(af, v1, o[half * 4 + 1].v);
        o[half * 4 + 2].v = wmma_bf16(af, v2, o[half * 4 + 2].v);
        o[half * 4 + 3].v = wmma_bf16(af, v3, o[half * 4 + 3].v);
      }
    }
  }

  // finalize: O /= l, store bf16 into [token][h*128+d]
#pragma unroll
  for (int i = 0; i < 8; ++i) {
    const float inv = 1.0f / lrow[i];
    const size_t row = (size_t)(b * SEQ + qb0 + wave * 16 + g * 8 + i);
#pragma unroll
    for (int d = 0; d < 8; ++d)
      O[row * HIDDEN_DIM + h * HEAD_DIM + d * 16 + c15] = f2bf(o[d].f[i] * inv);
  }
}

// ---------------- launcher ----------------
extern "C" void kernel_launch(void* const* d_in, const int* in_sizes, int n_in,
                              void* d_out, int out_size, void* d_ws, size_t ws_size,
                              hipStream_t stream) {
  (void)in_sizes; (void)n_in; (void)out_size; (void)ws_size;
  const float* hs  = (const float*)d_in[0];
  const float* wq  = (const float*)d_in[1];
  const float* wk  = (const float*)d_in[2];
  const float* wv  = (const float*)d_in[3];
  const float* wo  = (const float*)d_in[4];
  const float* wg  = (const float*)d_in[5];
  const float* wu  = (const float*)d_in[6];
  const float* wd  = (const float*)d_in[7];
  const float* ln1 = (const float*)d_in[8];
  const float* ln2 = (const float*)d_in[9];
  float* out = (float*)d_out;

  // workspace carve-up (~202 MB)
  unsigned short* xb    = (unsigned short*)d_ws;
  unsigned short* qb    = xb    + (size_t)MTOK * HIDDEN_DIM;
  unsigned short* kb    = qb    + (size_t)MTOK * HIDDEN_DIM;
  unsigned short* vb    = kb    + (size_t)MTOK * HIDDEN_DIM;
  unsigned short* attnb = vb    + (size_t)MTOK * HIDDEN_DIM;
  unsigned short* x2b   = attnb + (size_t)MTOK * HIDDEN_DIM;
  unsigned short* actb  = x2b   + (size_t)MTOK * HIDDEN_DIM;
  float*          hid2  = (float*)(actb + (size_t)MTOK * INTER_DIM);

  const dim3 blk(256);
  const dim3 gProj(HIDDEN_DIM / 128, MTOK / 128);
  const dim3 gMlp (INTER_DIM  / 128, MTOK / 128);

  // 1) x = rmsnorm(hidden, ln1) -> bf16
  rmsnorm_to_bf16_kernel<<<MTOK, blk, 0, stream>>>(hs, ln1, xb);
  // 2) Q,K,V projections (bf16 out)
  gemm_bf16_kernel<1><<<gProj, blk, 0, stream>>>(xb, wq, qb, nullptr,
                                                 MTOK, HIDDEN_DIM, HIDDEN_DIM);
  gemm_bf16_kernel<1><<<gProj, blk, 0, stream>>>(xb, wk, kb, nullptr,
                                                 MTOK, HIDDEN_DIM, HIDDEN_DIM);
  gemm_bf16_kernel<1><<<gProj, blk, 0, stream>>>(xb, wv, vb, nullptr,
                                                 MTOK, HIDDEN_DIM, HIDDEN_DIM);
  // 3) causal flash attention
  attn_kernel<<<dim3(SEQ / 128, NBATCH * NHEADS), blk, 0, stream>>>(qb, kb, vb, attnb);
  // 4) O projection + residual -> hid2 (f32)
  gemm_bf16_kernel<2><<<gProj, blk, 0, stream>>>(attnb, wo, hid2, hs,
                                                 MTOK, HIDDEN_DIM, HIDDEN_DIM);
  // 5) x2 = rmsnorm(hid2, ln2) -> bf16
  rmsnorm_to_bf16_kernel<<<MTOK, blk, 0, stream>>>(hid2, ln2, x2b);
  // 6) act = silu(x2*Wg) * (x2*Wu) -> bf16
  gemm_gateup_kernel<<<gMlp, blk, 0, stream>>>(x2b, wg, wu, actb,
                                               MTOK, INTER_DIM, HIDDEN_DIM);
  // 7) out = hid2 + act*Wd
  gemm_bf16_kernel<2><<<gProj, blk, 0, stream>>>(actb, wd, out, hid2,
                                                 MTOK, HIDDEN_DIM, INTER_DIM);
}
